// MultiHeadAttentionBlock_24996709663334
// MI455X (gfx1250) — compile-verified, matrix operand order checked by reading, not run on hardware
//
#include <hip/hip_runtime.h>

typedef __attribute__((ext_vector_type(16))) _Float16 v16h;
typedef __attribute__((ext_vector_type(8)))  _Float16 v8h;
typedef __attribute__((ext_vector_type(8)))  float    v8f;
typedef __attribute__((ext_vector_type(4)))  int      v4i;

#define DEV static __device__ __forceinline__

constexpr int DM   = 768;     // model dim
constexpr int SEQ  = 2048;    // sequence length
constexpr int NH   = 12;      // heads
constexpr int DK   = 64;      // head dim
constexpr int NROW = 2 * SEQ; // B*S = 4096 rows
constexpr int LDP  = 72;      // padded LDS row (halves): conflict-free b128 reads

// ---- CDNA5 async global->LDS copy (guarded; falls back to sync staging) -----
#if defined(__has_builtin) && __has_builtin(__builtin_amdgcn_global_load_async_to_lds_b128)
#define HAVE_ASYNC_LDS 1
DEV void async_cp16(_Float16* ldst, const _Float16* gsrc) {
  __builtin_amdgcn_global_load_async_to_lds_b128(
      (__attribute__((address_space(1))) v4i*)gsrc,
      (__attribute__((address_space(3))) v4i*)ldst,
      0, 0);
}
#if defined(__has_builtin) && __has_builtin(__builtin_amdgcn_s_wait_asynccnt)
#define ASYNC_WAIT(n) __builtin_amdgcn_s_wait_asynccnt(n)
#else
#define ASYNC_WAIT(n) asm volatile("s_wait_asynccnt " #n ::: "memory")
#endif
#else
#define HAVE_ASYNC_LDS 0
DEV void async_cp16(_Float16* ldst, const _Float16* gsrc) {
  *(v8h*)ldst = *(const v8h*)gsrc;   // sync fallback through VGPRs
}
#define ASYNC_WAIT(n) ((void)0)
#endif

// ---- fragment helpers -------------------------------------------------------
// CDNA5 16-bit A/B fragment (16x32 / 32x16), wave32:
//   lane l: row/col = l&15, g = l>>4
//   VGPR0..3 hold K = g*8 + 0..7, VGPR4..7 hold K = 16 + g*8 + 0..7
// => per lane: two contiguous 16-byte chunks at +0 and +16 halves.
DEV v16h frag_f16(const _Float16* p) {
  v8h lo = *(const v8h*)p;
  v8h hi = *(const v8h*)(p + 16);
  v16h r;
#pragma unroll
  for (int i = 0; i < 8; ++i) { r[i] = lo[i]; r[i + 8] = hi[i]; }
  return r;
}

DEV v8f wmma(v16h a, v16h b, v8f c) {
  return __builtin_amdgcn_wmma_f32_16x16x32_f16(false, a, false, b, (short)0, c,
                                                false, false);
}

DEV float rmax16(float v) {
  v = fmaxf(v, __shfl_xor(v, 1, 32));
  v = fmaxf(v, __shfl_xor(v, 2, 32));
  v = fmaxf(v, __shfl_xor(v, 4, 32));
  v = fmaxf(v, __shfl_xor(v, 8, 32));
  return v;
}
DEV float rsum16(float v) {
  v += __shfl_xor(v, 1, 32);
  v += __shfl_xor(v, 2, 32);
  v += __shfl_xor(v, 4, 32);
  v += __shfl_xor(v, 8, 32);
  return v;
}

// ---- elementwise f32 -> f16 (one pass; removes cvt from GEMM hot loops) -----
__global__ __launch_bounds__(256) void cvt16_kernel(const float* __restrict__ src,
                                                    _Float16* __restrict__ dst,
                                                    int n8) {
  const int i = blockIdx.x * 256 + threadIdx.x;
  if (i >= n8) return;
  const size_t off = (size_t)i * 8;
  v8f x = *(const v8f*)(src + off);
  v8h y;
#pragma unroll
  for (int j = 0; j < 8; ++j) y[j] = (_Float16)x[j];
  *(v8h*)(dst + off) = y;
}

// ---- QKV projection (all-f16 operands): Y = X @ W^T + b, head layout --------
// mode 0: Y[b][h][s][d]  (Q and K)        mode 1: Y[b][h][d][s]  (V transposed)
__global__ __launch_bounds__(128) void proj16_kernel(
    const _Float16* __restrict__ X, const _Float16* __restrict__ W,
    const float* __restrict__ bias, _Float16* __restrict__ Y, int mode) {
  const int lane = threadIdx.x & 31;
  const int wave = threadIdx.x >> 5;
  const int row  = lane & 15;
  const int g    = lane >> 4;
  const int rowbase = blockIdx.x * 64 + wave * 16;
  const int h       = blockIdx.y;          // 64-col tile == one head
  const int colbase = h * DK;

  v8f c[4] = {v8f{}, v8f{}, v8f{}, v8f{}};
  const _Float16* xrow = X + (size_t)(rowbase + row) * DM;

  for (int k0 = 0; k0 < DM; k0 += 32) {
    v16h a = frag_f16(xrow + k0 + g * 8);
#pragma unroll
    for (int t = 0; t < 4; ++t) {
      v16h b = frag_f16(W + (size_t)(colbase + t * 16 + row) * DM + k0 + g * 8);
      c[t] = wmma(a, b, c[t]);
    }
  }

  const int mrow0 = g * 8;
#pragma unroll
  for (int t = 0; t < 4; ++t) {
    const int d = t * 16 + row;
    const float bv = bias[colbase + d];
#pragma unroll
    for (int r = 0; r < 8; ++r) {
      const int grow = rowbase + mrow0 + r;
      const int bb = grow >> 11;
      const int s  = grow & (SEQ - 1);
      const float val = c[t][r] + bv;
      size_t off;
      if (mode == 0) off = ((size_t)(bb * NH + h) * SEQ + s) * DK + d;
      else           off = ((size_t)(bb * NH + h) * DK + d) * SEQ + s;
      Y[off] = (_Float16)val;
    }
  }
}

// ---- fused flash attention with async double-buffered LDS staging -----------
__global__ __launch_bounds__(128) void attn_kernel(
    const _Float16* __restrict__ Q, const _Float16* __restrict__ K,
    const _Float16* __restrict__ Vt, const int* __restrict__ mask,
    _Float16* __restrict__ AO) {
  __shared__ __align__(16) _Float16 sK[2][64 * LDP];  // K tile, [key][d] padded
  __shared__ __align__(16) _Float16 sV[2][64 * LDP];  // V^T tile, [d][key] padded
  __shared__ __align__(16) _Float16 sP[4][16 * LDP];  // per-wave P transpose

  const int tid  = threadIdx.x;
  const int lane = tid & 31;
  const int wave = tid >> 5;
  const int row  = lane & 15;
  const int g    = lane >> 4;
  const int h = blockIdx.y, b = blockIdx.z;
  const int bh = b * NH + h;
  const int qbase = blockIdx.x * 64 + wave * 16;

  const _Float16* Qh = Q  + (size_t)bh * SEQ * DK;
  const _Float16* Kh = K  + (size_t)bh * SEQ * DK;
  const _Float16* Vh = Vt + (size_t)bh * DK * SEQ;

  const _Float16* qrow = Qh + (size_t)(qbase + row) * DK;
  const v16h qf0 = frag_f16(qrow + 0  + g * 8);
  const v16h qf1 = frag_f16(qrow + 32 + g * 8);

  float m_i[8], l_i[8];
#pragma unroll
  for (int r = 0; r < 8; ++r) { m_i[r] = -3.0e38f; l_i[r] = 0.0f; }
  v8f o[4] = {v8f{}, v8f{}, v8f{}, v8f{}};

  const int mrow0 = g * 8;
  _Float16* pl = sP[wave];

  // stage one 64-key tile (8 KB K + 8 KB V) via 8 async b128 ops per thread
  auto stage = [&](int kb, int buf) {
#pragma unroll
    for (int i = 0; i < 4; ++i) {
      const int c  = tid + i * 128;       // 0..511 16-byte chunks
      const int r  = c >> 3;              // tile row 0..63
      const int ch = (c & 7) * 8;         // halves within row
      async_cp16(&sK[buf][r * LDP + ch], Kh + (size_t)(kb + r) * DK + ch);
      async_cp16(&sV[buf][r * LDP + ch], Vh + (size_t)r * SEQ + kb + ch);
    }
  };

  stage(0, 0);
  for (int kb = 0; kb < SEQ; kb += 64) {
    const int cur = (kb >> 6) & 1;
    if (kb + 64 < SEQ) { stage(kb + 64, cur ^ 1); ASYNC_WAIT(8); }
    else               { ASYNC_WAIT(0); }
    __syncthreads();

    const _Float16* kB = sK[cur];
    const _Float16* vB = sV[cur];

    // S = (Q K^T) * 1/8, 16q x 64k
    v8f c[4];
#pragma unroll
    for (int t = 0; t < 4; ++t) {
      const _Float16* krow = kB + (t * 16 + row) * LDP + g * 8;
      v8f z{};
      c[t] = wmma(qf0, frag_f16(krow), z);
      c[t] = wmma(qf1, frag_f16(krow + 32), c[t]);
    }
    // scale + mask
#pragma unroll
    for (int t = 0; t < 4; ++t) {
      const int kn = kb + t * 16 + row;
#pragma unroll
      for (int r = 0; r < 8; ++r) {
        const int qm = qbase + mrow0 + r;
        const int mv = mask[(size_t)qm * SEQ + kn];
        c[t][r] = (mv == 0) ? -1.0e9f : c[t][r] * 0.125f;
      }
    }
    // online softmax
    float mnew[8], alpha[8];
#pragma unroll
    for (int r = 0; r < 8; ++r) {
      float v = fmaxf(fmaxf(c[0][r], c[1][r]), fmaxf(c[2][r], c[3][r]));
      mnew[r]  = fmaxf(m_i[r], rmax16(v));
      alpha[r] = __expf(m_i[r] - mnew[r]);
      m_i[r]   = mnew[r];
    }
#pragma unroll
    for (int t = 0; t < 4; ++t)
#pragma unroll
      for (int r = 0; r < 8; ++r) c[t][r] = __expf(c[t][r] - mnew[r]);
#pragma unroll
    for (int r = 0; r < 8; ++r) {
      l_i[r] = l_i[r] * alpha[r] +
               rsum16(c[0][r] + c[1][r] + c[2][r] + c[3][r]);
      o[0][r] *= alpha[r]; o[1][r] *= alpha[r];
      o[2][r] *= alpha[r]; o[3][r] *= alpha[r];
    }
    // P: C-layout registers -> LDS -> A-layout fragments (per-wave buffer)
#pragma unroll
    for (int t = 0; t < 4; ++t)
#pragma unroll
      for (int r = 0; r < 8; ++r)
        pl[(mrow0 + r) * LDP + t * 16 + row] = (_Float16)c[t][r];
    asm volatile("s_wait_dscnt 0x0" ::: "memory");
    const v16h p0 = frag_f16(pl + row * LDP + 0  + g * 8);
    const v16h p1 = frag_f16(pl + row * LDP + 32 + g * 8);
    // O += P @ V
#pragma unroll
    for (int t = 0; t < 4; ++t) {
      const _Float16* vrow = vB + (t * 16 + row) * LDP + g * 8;
      o[t] = wmma(p0, frag_f16(vrow), o[t]);
      o[t] = wmma(p1, frag_f16(vrow + 32), o[t]);
    }
    __syncthreads();  // all waves done reading before this buffer is re-staged
  }

  // epilogue: O / l, scatter to [B,S,H*DK] f16
  float inv[8];
#pragma unroll
  for (int r = 0; r < 8; ++r) inv[r] = 1.0f / l_i[r];
#pragma unroll
  for (int t = 0; t < 4; ++t) {
    const int col = h * DK + t * 16 + row;
#pragma unroll
    for (int r = 0; r < 8; ++r) {
      const int grow = b * SEQ + qbase + mrow0 + r;
      AO[(size_t)grow * DM + col] = (_Float16)(o[t][r] * inv[r]);
    }
  }
}

// ---- output projection: f32 out = AO(f16) @ Wo^T(f16) + bo ------------------
__global__ __launch_bounds__(128) void oproj16_kernel(
    const _Float16* __restrict__ X16, const _Float16* __restrict__ W,
    const float* __restrict__ bias, float* __restrict__ Y) {
  const int lane = threadIdx.x & 31;
  const int wave = threadIdx.x >> 5;
  const int row  = lane & 15;
  const int g    = lane >> 4;
  const int rowbase = blockIdx.x * 64 + wave * 16;
  const int colbase = blockIdx.y * 64;

  v8f c[4] = {v8f{}, v8f{}, v8f{}, v8f{}};
  const _Float16* xrow = X16 + (size_t)(rowbase + row) * DM;

  for (int k0 = 0; k0 < DM; k0 += 32) {
    v16h a = frag_f16(xrow + k0 + g * 8);
#pragma unroll
    for (int t = 0; t < 4; ++t) {
      v16h b = frag_f16(W + (size_t)(colbase + t * 16 + row) * DM + k0 + g * 8);
      c[t] = wmma(a, b, c[t]);
    }
  }
  const int mrow0 = g * 8;
#pragma unroll
  for (int t = 0; t < 4; ++t) {
    const int n = colbase + t * 16 + row;
    const float bv = bias[n];
#pragma unroll
    for (int r = 0; r < 8; ++r)
      Y[(size_t)(rowbase + mrow0 + r) * DM + n] = c[t][r] + bv;
  }
}

// ---- launch -----------------------------------------------------------------
extern "C" void kernel_launch(void* const* d_in, const int* in_sizes, int n_in,
                              void* d_out, int out_size, void* d_ws, size_t ws_size,
                              hipStream_t stream) {
  const float* q    = (const float*)d_in[0];
  const float* k    = (const float*)d_in[1];
  const float* v    = (const float*)d_in[2];
  const int*   mask = (const int*)  d_in[3];
  const float* w_q  = (const float*)d_in[4];
  const float* b_q  = (const float*)d_in[5];
  const float* w_k  = (const float*)d_in[6];
  const float* b_k  = (const float*)d_in[7];
  const float* w_v  = (const float*)d_in[8];
  const float* b_v  = (const float*)d_in[9];
  const float* w_o  = (const float*)d_in[10];
  const float* b_o  = (const float*)d_in[11];

  const size_t nXW = (size_t)NROW * DM;   // 4096*768
  const size_t nWW = (size_t)DM * DM;     // 768*768
  _Float16* p = (_Float16*)d_ws;
  _Float16* xq16 = p; p += nXW;
  _Float16* xk16 = p; p += nXW;
  _Float16* xv16 = p; p += nXW;
  _Float16* wq16 = p; p += nWW;
  _Float16* wk16 = p; p += nWW;
  _Float16* wv16 = p; p += nWW;
  _Float16* wo16 = p; p += nWW;
  _Float16* q16  = p; p += nXW;
  _Float16* k16  = p; p += nXW;
  _Float16* vt16 = p; p += nXW;
  _Float16* ao16 = p; p += nXW;

  const dim3 blk(128);
  const int nx8 = (int)(nXW / 8), nw8 = (int)(nWW / 8);
  cvt16_kernel<<<(nx8 + 255) / 256, 256, 0, stream>>>(q,   xq16, nx8);
  cvt16_kernel<<<(nx8 + 255) / 256, 256, 0, stream>>>(k,   xk16, nx8);
  cvt16_kernel<<<(nx8 + 255) / 256, 256, 0, stream>>>(v,   xv16, nx8);
  cvt16_kernel<<<(nw8 + 255) / 256, 256, 0, stream>>>(w_q, wq16, nw8);
  cvt16_kernel<<<(nw8 + 255) / 256, 256, 0, stream>>>(w_k, wk16, nw8);
  cvt16_kernel<<<(nw8 + 255) / 256, 256, 0, stream>>>(w_v, wv16, nw8);
  cvt16_kernel<<<(nw8 + 255) / 256, 256, 0, stream>>>(w_o, wo16, nw8);

  const dim3 pg(NROW / 64, DM / 64);      // 64 x 12
  proj16_kernel<<<pg, blk, 0, stream>>>(xq16, wq16, b_q, q16, 0);
  proj16_kernel<<<pg, blk, 0, stream>>>(xk16, wk16, b_k, k16, 0);
  proj16_kernel<<<pg, blk, 0, stream>>>(xv16, wv16, b_v, vt16, 1);

  attn_kernel<<<dim3(SEQ / 64, NH, 2), blk, 0, stream>>>(q16, k16, vt16, mask, ao16);

  oproj16_kernel<<<pg, blk, 0, stream>>>(ao16, wo16, b_o, (float*)d_out);
}
